// VnMriReconCell_16149077033819
// MI455X (gfx1250) — compile-verified, matrix operand order checked by reading, not run on hardware
//
#include <hip/hip_runtime.h>

typedef __attribute__((ext_vector_type(16))) _Float16 v16h;
typedef __attribute__((ext_vector_type(8)))  float    v8f;
typedef unsigned short u16;

#define DEVI __device__ __forceinline__

constexpr int BN = 4, BC = 15, H = 320, W = 320;
constexpr int F = 32, KS = 11, PADR = 11, CPAD = 5, KRBF = 31;
constexpr int HP    = H + 2 * PADR;      // 342 (conv output grid)
constexpr int HP2   = HP + 2 * CPAD;     // 352 (zero-halo storage grid)
constexpr int NPIX  = H * W;             // 102400
constexpr int NPPAD = HP * HP;           // 116964
constexpr int NPPAD2= HP2 * HP2;         // 123904
constexpr int IMGS  = BN * BC;           // 60
constexpr int KXP   = 16;                // kernel row padded 11 -> 16
constexpr int NROW  = 2 * KS;            // 22 (c,ky) rows; K = 352
constexpr int NCH2  = NROW / 2;          // 11 K-chunks of 32
constexpr int NTAP  = KS * KS;           // 121
constexpr int MT_CONV = (BN * NPPAD) / 16; // 29241 M-tiles of 16 pixels

// ---- workspace layout (bytes) ----
constexpr size_t AL(size_t x) { return (x + 255) & ~(size_t)255; }
constexpr size_t O_UPAD = 0;                 // 352^2 halo + 16-elem zero guard
constexpr size_t O_WP1  = AL(O_UPAD + ((size_t)BN * 2 * NPPAD2 + 16) * sizeof(u16));
constexpr size_t O_WP2  = AL(O_WP1 + (size_t)NCH2 * 2 * 32 * 16 * sizeof(u16));
constexpr size_t O_FUK  = AL(O_WP2 + (size_t)NTAP * 32 * 16 * sizeof(u16));   // 352^2 halo
constexpr size_t O_RU   = AL(O_FUK + (size_t)BN * NPPAD2 * F * sizeof(u16));
constexpr size_t O_FR   = AL(O_RU + (size_t)BN * NPPAD * 2 * sizeof(float));
constexpr size_t O_FI   = AL(O_FR + (size_t)NPIX * sizeof(u16));
constexpr size_t O_FIN  = AL(O_FI + (size_t)NPIX * sizeof(u16));
constexpr size_t O_XR   = AL(O_FIN + (size_t)NPIX * sizeof(u16));
constexpr size_t O_XI   = AL(O_XR + (size_t)IMGS * NPIX * sizeof(u16));
constexpr size_t O_YR   = AL(O_XI + (size_t)IMGS * NPIX * sizeof(u16));
constexpr size_t O_YI   = AL(O_YR + (size_t)IMGS * NPIX * sizeof(u16));
constexpr size_t O_AR   = AL(O_YI + (size_t)IMGS * NPIX * sizeof(u16));
constexpr size_t O_AI   = AL(O_AR + (size_t)IMGS * NPIX * sizeof(u16));
constexpr size_t O_ACC  = AL(O_AI + (size_t)IMGS * NPIX * sizeof(u16));

// ---- fp16 helpers ----
DEVI u16 f2h(float f) {
    _Float16 h = (_Float16)f;
    return __builtin_bit_cast(unsigned short, h);
}
DEVI float h2f(u16 s) {
    _Float16 h = __builtin_bit_cast(_Float16, (unsigned short)s);
    return (float)h;
}

union FragH { u16 s[16]; uint4 q[2]; v16h v; };
union Pack8 { u16 s[8]; uint4 q; };

DEVI v8f wmma_f16(v16h a, v16h b, v8f c) {
    return __builtin_amdgcn_wmma_f32_16x16x32_f16(false, a, false, b, (short)0, c,
                                                  false, false);
}

// ---- 1: pack conv weights into exact B-fragment lane order ----
// wp1: [chunk(11)][ntile(2)][lane(32)][16]; K = c*176 + ky*16 + kx (kx>=11 -> 0)
// wp2: [tap(121)][lane(32)][16]; per tap K-chunk = feature f, N = channel
__global__ void k_pack(const float* __restrict__ ck, u16* wp1, u16* wp2) {
    int idx = blockIdx.x * 256 + threadIdx.x;
    if (idx < NCH2 * 2 * 32 * 16) {
        int e = idx & 15, lane = (idx >> 4) & 31, nt = (idx >> 9) & 1, cb = idx >> 10;
        int hl = lane >> 4, lr = lane & 15;
        int kl = (e < 8) ? hl * 8 + e : 16 + hl * 8 + (e - 8);
        int row = 2 * cb + (kl >> 4);          // (c,ky) row index 0..21
        int kx = kl & 15;
        int c = row / KS, ky = row % KS;
        float v = 0.f;
        if (kx < KS) {
            int fch = nt * 16 + lr;
            v = ck[(((fch * KS) + ky) * KS + kx) * 2 + c];
        }
        wp1[idx] = f2h(v);
    }
    int j = idx - NCH2 * 2 * 32 * 16;
    if (j >= 0 && j < NTAP * 32 * 16) {
        int e = j & 15, lane = (j >> 4) & 31, tap = j >> 9;
        int hl = lane >> 4, lr = lane & 15;
        int fch = (e < 8) ? hl * 8 + e : 16 + hl * 8 + (e - 8); // feature
        int dy = tap / KS, dx = tap % KS;
        float v = 0.f;
        if (lr < 2)  // N col 0=real, 1=imag; flip+transpose baked in
            v = ck[((fch * KS + (KS - 1 - dy)) * KS + (KS - 1 - dx)) * 2 + lr];
        wp2[j] = f2h(v);
    }
}

// ---- 2: centered DFT matrix (symmetric): F = cos - i sin, scaled 1/sqrt(320) ----
__global__ void k_dft(u16* fr, u16* fi, u16* fin) {
    int idx = blockIdx.x * 256 + threadIdx.x;
    if (idx >= NPIX) return;
    int m = idx / W - W / 2, n = idx % W - W / 2;
    int t = (m * n) % W; if (t < 0) t += W;
    float ang = -6.283185307179586f * (float)t / (float)W;
    float sv, cv; __sincosf(ang, &sv, &cv);
    const float nrm = 0.05590169943749474f; // 1/sqrt(320)
    fr[idx]  = f2h(cv * nrm);
    fi[idx]  = f2h(sv * nrm);
    fin[idx] = f2h(-sv * nrm);
}

// ---- 3: reflect pad into 352^2 zero-halo layout (N,2,352,352) fp16 + guard ----
DEVI int refl(int i, int n) { if (i < 0) i = -i; if (i >= n) i = 2 * n - 2 - i; return i; }
__global__ void k_pad(const float* __restrict__ ut, u16* up) {
    int idx = blockIdx.x * 256 + threadIdx.x;
    if (idx >= BN * 2 * NPPAD2 + 16) return;
    if (idx >= BN * 2 * NPPAD2) { up[idx] = 0; return; }   // zero guard tail
    int n = idx / (2 * NPPAD2), r = idx % (2 * NPPAD2), c = r / NPPAD2, p = r % NPPAD2;
    int y2 = p / HP2, x2 = p % HP2;
    float v = 0.f;
    if ((unsigned)(y2 - CPAD) < (unsigned)HP && (unsigned)(x2 - CPAD) < (unsigned)HP) {
        int ys = refl(y2 - CPAD - PADR, H), xs = refl(x2 - CPAD - PADR, W);
        v = ut[((size_t)(n * H + ys) * W + xs) * 2 + c];
    }
    up[idx] = f2h(v);
}

// ---- 3b: zero the 5-px halo ring of fuk (materialized conv zero-padding) ----
__global__ void k_zhalo(u16* __restrict__ fuk) {
    int idx = blockIdx.x * 256 + threadIdx.x;
    if (idx >= BN * NPPAD2) return;
    int p = idx % NPPAD2;
    int y = p / HP2, x = p % HP2;
    if (y < CPAD || y >= HP2 - CPAD || x < CPAD || x >= HP2 - CPAD) {
        uint4 z = make_uint4(0u, 0u, 0u, 0u);
        uint4* d = (uint4*)(fuk + (size_t)idx * F);
        d[0] = z; d[1] = z; d[2] = z; d[3] = z;
    }
}

// ---- 4: implicit-GEMM conv with row16-padded K: vector A loads, no masks ----
__global__ void __launch_bounds__(256) k_conv_rbf(
    const u16* __restrict__ up, const u16* __restrict__ wp1,
    const float* __restrict__ w, const float* __restrict__ mu,
    const float* __restrict__ sigma, u16* __restrict__ fuk) {
    __shared__ __align__(16) u16 swp[NCH2 * 2 * 32 * 16]; // 22KB weight fragments
    {   // async global->LDS stage of pre-swizzled weight fragments (ASYNCcnt path)
        unsigned lds0 = (unsigned)(uintptr_t)&swp[0];
        for (int i = threadIdx.x; i < NCH2 * 2 * 32 * 16 / 8; i += 256) {
            unsigned loff = lds0 + (unsigned)i * 16u;
            unsigned goff = (unsigned)i * 16u;
            asm volatile("global_load_async_to_lds_b128 %0, %1, %2"
                         :: "v"(loff), "v"(goff), "s"(wp1) : "memory");
        }
        asm volatile("s_wait_asynccnt 0" ::: "memory");
    }
    __syncthreads();
    int wave = threadIdx.x >> 5, lane = threadIdx.x & 31;
    int hl = lane >> 4, lr = lane & 15;
    int tile = blockIdx.x * 8 + wave;
    if (tile >= MT_CONV) return;            // wave-uniform exit
    int g0 = tile * 16;
    int n0 = g0 / NPPAD, p0 = g0 - n0 * NPPAD;
    int pl = p0 + lr, nl = n0;
    if (pl >= NPPAD) { pl -= NPPAD; nl++; }
    int y = pl / HP, x = pl - (pl / HP) * HP;
    // lane's A base: halo row y, col x, plus this half's kx offset
    const u16* pbase = up + (size_t)nl * 2 * NPPAD2 + (size_t)y * HP2 + x + hl * 8;
    v8f acc0 = {}; v8f acc1 = {};
#pragma unroll
    for (int cb = 0; cb < NCH2; ++cb) {
        int r0 = 2 * cb, r1 = 2 * cb + 1;          // compile-time (c,ky) rows
        int c0 = r0 / KS, ky0 = r0 % KS;
        int c1 = r1 / KS, ky1 = r1 % KS;
        FragH a;                                   // two 16B row-segment loads
        __builtin_memcpy(&a.s[0], pbase + (size_t)c0 * NPPAD2 + (size_t)ky0 * HP2, 16);
        __builtin_memcpy(&a.s[8], pbase + (size_t)c1 * NPPAD2 + (size_t)ky1 * HP2, 16);
        FragH b0, b1;
        b0.q[0] = *(const uint4*)&swp[((cb * 2 + 0) * 32 + lane) * 16];
        b0.q[1] = *(const uint4*)&swp[((cb * 2 + 0) * 32 + lane) * 16 + 8];
        b1.q[0] = *(const uint4*)&swp[((cb * 2 + 1) * 32 + lane) * 16];
        b1.q[1] = *(const uint4*)&swp[((cb * 2 + 1) * 32 + lane) * 16 + 8];
        acc0 = wmma_f16(a.v, b0.v, acc0);
        acc1 = wmma_f16(a.v, b1.v, acc1);
    }
    // RBF epilogue: lane owns feats lr & lr+16, pixels r + 8*hl
    float inv2s2 = 1.f / (2.f * sigma[0] * sigma[0]);
    int f0 = lr, f1 = lr + 16;
    float r0a[8], r1a[8];
#pragma unroll
    for (int r = 0; r < 8; ++r) { r0a[r] = 0.f; r1a[r] = 0.f; }
    for (int i = 0; i < KRBF; ++i) {
        float mi = mu[i];
        float w0 = w[f0 * KRBF + i], w1 = w[f1 * KRBF + i];
#pragma unroll
        for (int r = 0; r < 8; ++r) {
            float d0 = acc0[r] - mi, d1 = acc1[r] - mi;
            r0a[r] += w0 * __expf(-d0 * d0 * inv2s2);
            r1a[r] += w1 * __expf(-d1 * d1 * inv2s2);
        }
    }
#pragma unroll
    for (int r = 0; r < 8; ++r) {
        int pr = p0 + r + 8 * hl, nr = n0;
        if (pr >= NPPAD) { pr -= NPPAD; nr++; }
        int yr = pr / HP, xr = pr - yr * HP;
        size_t base = ((size_t)nr * NPPAD2 + (size_t)(yr + CPAD) * HP2 + (xr + CPAD)) * F;
        fuk[base + f0] = f2h(r0a[r]);
        fuk[base + f1] = f2h(r1a[r]);
    }
}

// ---- 5: implicit-GEMM transposed conv: pure loads + WMMA (no bounds checks) ----
__global__ void __launch_bounds__(256) k_convT(
    const u16* __restrict__ fuk, const u16* __restrict__ wp2,
    float* __restrict__ ru) {
    int wave = threadIdx.x >> 5, lane = threadIdx.x & 31;
    int hl = lane >> 4, lr = lane & 15;
    int tile = blockIdx.x * 8 + wave;
    if (tile >= MT_CONV) return;
    int g0 = tile * 16;
    int n0 = g0 / NPPAD, p0 = g0 - n0 * NPPAD;
    int pl = p0 + lr, nl = n0;
    if (pl >= NPPAD) { pl -= NPPAD; nl++; }
    int y = pl / HP, x = pl - (pl / HP) * HP;
    const u16* pb = fuk + ((size_t)nl * NPPAD2 + (size_t)y * HP2 + x) * F + hl * 8;
    const u16* wrow = wp2 + (size_t)lane * 16;
    v8f acc = {};
    for (int dy = 0; dy < KS; ++dy) {
        const u16* rowp = pb + (size_t)dy * HP2 * F;
        for (int dx = 0; dx < KS; ++dx) {
            FragH a;
            const u16* ap = rowp + (size_t)dx * F;
            a.q[0] = *(const uint4*)ap;
            a.q[1] = *(const uint4*)(ap + 16);
            FragH b;
            const u16* bp = wrow + (size_t)(dy * KS + dx) * 512;
            b.q[0] = *(const uint4*)bp;
            b.q[1] = *(const uint4*)(bp + 8);
            acc = wmma_f16(a.v, b.v, acc);
        }
    }
    if (lr < 2) {                            // only N cols 0,1 are real/imag
#pragma unroll
        for (int r = 0; r < 8; ++r) {
            int pr = p0 + r + 8 * hl, nr = n0;
            if (pr >= NPPAD) { pr -= NPPAD; nr++; }
            ru[((size_t)nr * NPPAD + pr) * 2 + lr] = acc[r];
        }
    }
}

// ---- 6: X = u * coil_sens (complex), fp16 planes ----
__global__ void k_X(const float* __restrict__ ut, const float* __restrict__ cs,
                    u16* xr, u16* xi) {
    int idx = blockIdx.x * 256 + threadIdx.x;
    if (idx >= IMGS * NPIX) return;
    int img = idx / NPIX, p = idx % NPIX, n = img / BC;
    float ur = ut[((size_t)n * NPIX + p) * 2], ui = ut[((size_t)n * NPIX + p) * 2 + 1];
    float cr = cs[(size_t)idx * 2], ci = cs[(size_t)idx * 2 + 1];
    xr[idx] = f2h(ur * cr - ui * ci);
    xi[idx] = f2h(ur * ci + ui * cr);
}

// ---- 7: complex GEMM, C = (A*B)^T; one ni per block, B staged in LDS async ----
__global__ void __launch_bounds__(320) k_cgemmT(
    const u16* __restrict__ Ar, const u16* __restrict__ Ai,
    const u16* __restrict__ Br, const u16* __restrict__ Bi,
    const u16* __restrict__ Bin, u16* __restrict__ Cr, u16* __restrict__ Ci) {
    __shared__ __align__(16) u16 sB[3 * 16 * W];   // 30KB: 16 rows x 320 of Br/Bi/Bin
    int ni = blockIdx.x;
    size_t ib = (size_t)blockIdx.y * NPIX;
    {   // async global->LDS stage of the three B row-blocks
        unsigned lds0 = (unsigned)(uintptr_t)&sB[0];
#pragma unroll
        for (int pl = 0; pl < 3; ++pl) {
            const u16* gp = (pl == 0) ? Br : (pl == 1) ? Bi : Bin;
            for (int i = threadIdx.x; i < 16 * W / 8; i += 320) {  // 640 b128 per plane
                unsigned loff = lds0 + (unsigned)(pl * 16 * W + i * 8) * 2u;
                unsigned goff = (unsigned)(ni * 16 * W + i * 8) * 2u;
                asm volatile("global_load_async_to_lds_b128 %0, %1, %2"
                             :: "v"(loff), "v"(goff), "s"(gp) : "memory");
            }
        }
        asm volatile("s_wait_asynccnt 0" ::: "memory");
    }
    __syncthreads();
    int wave = threadIdx.x >> 5, lane = threadIdx.x & 31;
    int hl = lane >> 4, lr = lane & 15;
    int mi0 = wave, mi1 = wave + 10;               // 10 waves x 2 M-tiles = 20
    const u16* a0r = Ar + ib + (size_t)(mi0 * 16 + lr) * W;
    const u16* a0i = Ai + ib + (size_t)(mi0 * 16 + lr) * W;
    const u16* a1r = Ar + ib + (size_t)(mi1 * 16 + lr) * W;
    const u16* a1i = Ai + ib + (size_t)(mi1 * 16 + lr) * W;
    const u16* sbr = &sB[0 * 16 * W + lr * W];
    const u16* sbi = &sB[1 * 16 * W + lr * W];
    const u16* sbn = &sB[2 * 16 * W + lr * W];
    v8f cr0 = {}, ci0 = {}, cr1 = {}, ci1 = {};
    for (int cb = 0; cb < 10; ++cb) {
        int k0 = cb * 32 + hl * 8;
        FragH br_, bi_, bn_;
        br_.q[0] = *(const uint4*)(sbr + k0); br_.q[1] = *(const uint4*)(sbr + k0 + 16);
        bi_.q[0] = *(const uint4*)(sbi + k0); bi_.q[1] = *(const uint4*)(sbi + k0 + 16);
        bn_.q[0] = *(const uint4*)(sbn + k0); bn_.q[1] = *(const uint4*)(sbn + k0 + 16);
        FragH f0r, f0i, f1r, f1i;
        f0r.q[0] = *(const uint4*)(a0r + k0); f0r.q[1] = *(const uint4*)(a0r + k0 + 16);
        f0i.q[0] = *(const uint4*)(a0i + k0); f0i.q[1] = *(const uint4*)(a0i + k0 + 16);
        f1r.q[0] = *(const uint4*)(a1r + k0); f1r.q[1] = *(const uint4*)(a1r + k0 + 16);
        f1i.q[0] = *(const uint4*)(a1i + k0); f1i.q[1] = *(const uint4*)(a1i + k0 + 16);
        if (cb < 9) {
            __builtin_prefetch(a0r + k0 + 32, 0, 0);
            __builtin_prefetch(a1r + k0 + 32, 0, 0);
        }
        cr0 = wmma_f16(f0r.v, br_.v, cr0);  cr0 = wmma_f16(f0i.v, bn_.v, cr0);
        ci0 = wmma_f16(f0r.v, bi_.v, ci0);  ci0 = wmma_f16(f0i.v, br_.v, ci0);
        cr1 = wmma_f16(f1r.v, br_.v, cr1);  cr1 = wmma_f16(f1i.v, bn_.v, cr1);
        ci1 = wmma_f16(f1r.v, bi_.v, ci1);  ci1 = wmma_f16(f1i.v, br_.v, ci1);
    }
    // transposed store: lane's 8 elements are contiguous -> one b128 per plane/tile
    Pack8 pr0, pi0, pr1, pi1;
#pragma unroll
    for (int r = 0; r < 8; ++r) {
        pr0.s[r] = f2h(cr0[r]); pi0.s[r] = f2h(ci0[r]);
        pr1.s[r] = f2h(cr1[r]); pi1.s[r] = f2h(ci1[r]);
    }
    size_t rowo = ib + (size_t)(ni * 16 + lr) * W;
    *(uint4*)(Cr + rowo + mi0 * 16 + 8 * hl) = pr0.q;
    *(uint4*)(Ci + rowo + mi0 * 16 + 8 * hl) = pi0.q;
    *(uint4*)(Cr + rowo + mi1 * 16 + 8 * hl) = pr1.q;
    *(uint4*)(Ci + rowo + mi1 * 16 + 8 * hl) = pi1.q;
}

// ---- 8: R = mask * (Au - f) ----
__global__ void k_resid(const u16* __restrict__ aur, const u16* __restrict__ aui,
                        const float* __restrict__ f, const float* __restrict__ msk,
                        u16* rr, u16* ri) {
    int idx = blockIdx.x * 256 + threadIdx.x;
    if (idx >= IMGS * NPIX) return;
    int n = (idx / NPIX) / BC, p = idx % NPIX;
    float m = msk[(size_t)n * NPIX + p];
    rr[idx] = f2h(m * (h2f(aur[idx]) - f[(size_t)idx * 2]));
    ri[idx] = f2h(m * (h2f(aui[idx]) - f[(size_t)idx * 2 + 1]));
}

// ---- 9: At = sum_coils ifft_result * conj(coil) ----
__global__ void k_accum(const u16* __restrict__ atr, const u16* __restrict__ ati,
                        const float* __restrict__ cs, float* __restrict__ acc) {
    int idx = blockIdx.x * 256 + threadIdx.x;
    if (idx >= BN * NPIX) return;
    int n = idx / NPIX, p = idx % NPIX;
    float sr = 0.f, si = 0.f;
    for (int c = 0; c < BC; ++c) {
        size_t b = ((size_t)(n * BC + c)) * NPIX + p;
        float ar = h2f(atr[b]), ai = h2f(ati[b]);
        float cr = cs[b * 2], ci = cs[b * 2 + 1];
        sr += ar * cr + ai * ci;
        si += ai * cr - ar * ci;
    }
    acc[(size_t)idx * 2]     = sr;
    acc[(size_t)idx * 2 + 1] = si;
}

// ---- 10: out = u - Ru(crop)/F - lamb*Du ----
__global__ void k_final(const float* __restrict__ ut, const float* __restrict__ ru,
                        const float* __restrict__ acc, const float* __restrict__ lamb,
                        float* __restrict__ out) {
    int idx = blockIdx.x * 256 + threadIdx.x;
    if (idx >= BN * NPIX) return;
    int n = idx / NPIX, p = idx % NPIX, y = p / W, x = p % W;
    size_t rp = ((size_t)n * NPPAD + (size_t)(y + PADR) * HP + (x + PADR)) * 2;
    float l = lamb[0];
    const float invF = 1.0f / (float)F;
    out[(size_t)idx * 2]     = ut[(size_t)idx * 2]     - ru[rp] * invF     - l * acc[(size_t)idx * 2];
    out[(size_t)idx * 2 + 1] = ut[(size_t)idx * 2 + 1] - ru[rp + 1] * invF - l * acc[(size_t)idx * 2 + 1];
}

extern "C" void kernel_launch(void* const* d_in, const int* in_sizes, int n_in,
                              void* d_out, int out_size, void* d_ws, size_t ws_size,
                              hipStream_t stream) {
    const float* ut   = (const float*)d_in[0];
    const float* f    = (const float*)d_in[1];
    const float* cs   = (const float*)d_in[2];
    const float* msk  = (const float*)d_in[3];
    const float* ck   = (const float*)d_in[4];
    const float* w    = (const float*)d_in[5];
    const float* mu   = (const float*)d_in[6];
    const float* sig  = (const float*)d_in[7];
    const float* lamb = (const float*)d_in[8];
    float* out = (float*)d_out;
    char* ws = (char*)d_ws;

    u16* upad = (u16*)(ws + O_UPAD);
    u16* wp1  = (u16*)(ws + O_WP1);
    u16* wp2  = (u16*)(ws + O_WP2);
    u16* fuk  = (u16*)(ws + O_FUK);
    float* ruf = (float*)(ws + O_RU);
    u16* fr  = (u16*)(ws + O_FR);
    u16* fi  = (u16*)(ws + O_FI);
    u16* fin = (u16*)(ws + O_FIN);
    u16* xr  = (u16*)(ws + O_XR);
    u16* xi  = (u16*)(ws + O_XI);
    u16* yr  = (u16*)(ws + O_YR);
    u16* yi  = (u16*)(ws + O_YI);
    u16* ar  = (u16*)(ws + O_AR);
    u16* ai  = (u16*)(ws + O_AI);
    float* atacc = (float*)(ws + O_ACC);

    dim3 b(256);
    // regularizer path
    k_pack<<<286, b, 0, stream>>>(ck, wp1, wp2);
    k_dft<<<(NPIX + 255) / 256, b, 0, stream>>>(fr, fi, fin);
    k_pad<<<(BN * 2 * NPPAD2 + 16 + 255) / 256, b, 0, stream>>>(ut, upad);
    k_zhalo<<<(BN * NPPAD2 + 255) / 256, b, 0, stream>>>(fuk);
    k_conv_rbf<<<(MT_CONV + 7) / 8, b, 0, stream>>>(upad, wp1, w, mu, sig, fuk);
    k_convT<<<(MT_CONV + 7) / 8, b, 0, stream>>>(fuk, wp2, ruf);
    // data-consistency path: fft2c = G(G(X)) with B = F; ifft2c uses conj(F)
    k_X<<<(IMGS * NPIX + 255) / 256, b, 0, stream>>>(ut, cs, xr, xi);
    dim3 gg(20, IMGS);
    dim3 bg(320);
    k_cgemmT<<<gg, bg, 0, stream>>>(xr, xi, fr, fi, fin, yr, yi);
    k_cgemmT<<<gg, bg, 0, stream>>>(yr, yi, fr, fi, fin, ar, ai);
    k_resid<<<(IMGS * NPIX + 255) / 256, b, 0, stream>>>(ar, ai, f, msk, xr, xi);
    k_cgemmT<<<gg, bg, 0, stream>>>(xr, xi, fr, fin, fi, yr, yi);   // conj(F)
    k_cgemmT<<<gg, bg, 0, stream>>>(yr, yi, fr, fin, fi, ar, ai);
    k_accum<<<(BN * NPIX + 255) / 256, b, 0, stream>>>(ar, ai, cs, atacc);
    k_final<<<(BN * NPIX + 255) / 256, b, 0, stream>>>(ut, ruf, atacc, lamb, out);
    (void)in_sizes; (void)n_in; (void)out_size; (void)ws_size;
}
